// GatedGCNAggregate_3942779978058
// MI455X (gfx1250) — compile-verified, matrix operand order checked by reading, not run on hardware
//
#include <hip/hip_runtime.h>
#include <math.h>

#define DH 256
#define GEMM_TR 128   // block tile rows
#define GEMM_TC 64    // block tile cols
#define KCH 32        // K chunk per WMMA step

typedef __attribute__((ext_vector_type(16))) __bf16 v16bf_t;
typedef __attribute__((ext_vector_type(8)))  float  v8f_t;

union BFrag { v16bf_t v; unsigned short u[16]; uint4 q[2]; };

__device__ __forceinline__ unsigned short f2bf(float f) {
  unsigned int u = __float_as_uint(f);
  u += 0x7FFFu + ((u >> 16) & 1u);   // round-to-nearest-even
  return (unsigned short)(u >> 16);
}

// ---------------------------------------------------------------------------
// Pre-swizzle a weight matrix W[K][256] (f32) into the exact WMMA B-fragment
// layout, bf16:  out[((kch*256 + col)*2 + half)*16 + j] = bf16(W[k][col])
// with k = kch*32 + (j/8)*16 + half*8 + (j%8).
// A lane (lr,half) of col-tile ct then reads its 16-elt fragment as two
// contiguous b128 loads; a wave's whole fragment is a contiguous 1KB block.
// ---------------------------------------------------------------------------
__global__ __launch_bounds__(256) void pack_weight_bf16(
    const float* __restrict__ W, unsigned short* __restrict__ out, int K)
{
  long i = (long)blockIdx.x * 256 + threadIdx.x;
  long total = (long)K * DH;
  if (i >= total) return;
  int j    = (int)(i & 15);
  int half = (int)((i >> 4) & 1);
  int col  = (int)((i >> 5) & 255);
  int kch  = (int)(i >> 13);
  int k = kch * KCH + ((j >> 3) << 4) + (half << 3) + (j & 7);
  out[i] = f2bf(W[(size_t)k * DH + col]);
}

// ---------------------------------------------------------------------------
// C[N x 256] = A[N x K](f32 -> bf16 via LDS) @ Bp(packed bf16) + bias
// Block = 256 threads = 8 waves. Block tile 128x64, wave tile 16x64.
// B fragments come straight from L2 (weights are 128KB, L2 is 192MB).
// ---------------------------------------------------------------------------
__global__ __launch_bounds__(256) void gemm_wmma_bf16(
    const float* __restrict__ A, const unsigned short* __restrict__ Bp,
    const float* __restrict__ bias, float* __restrict__ C, int N, int K)
{
  __shared__ unsigned short Alds[GEMM_TR][KCH];   // 8 KB

  const int tid  = threadIdx.x;
  const int wave = tid >> 5;
  const int lane = tid & 31;
  const int half = lane >> 4;      // k-half selector (fragment layout)
  const int lr   = lane & 15;      // row (A) / col (B,C) within 16

  const int rowBase = blockIdx.x * GEMM_TR;
  const int colBase = blockIdx.y * GEMM_TC;

  v8f_t acc[4];
#pragma unroll
  for (int t = 0; t < 4; ++t)
#pragma unroll
    for (int r = 0; r < 8; ++r) acc[t][r] = 0.0f;

  for (int kc = 0; kc < K; kc += KCH) {
    // ---- B fragments: 2 contiguous b128 loads per tile, no LDS ----
    BFrag bfr[4];
#pragma unroll
    for (int ct = 0; ct < 4; ++ct) {
      const uint4* bp = (const uint4*)(Bp +
          ((((size_t)(kc >> 5) * 256) + (size_t)(colBase + ct * 16 + lr)) * 2 + half) * 16);
      bfr[ct].q[0] = bp[0];
      bfr[ct].q[1] = bp[1];
    }

    // ---- A stage: 128x32 f32 -> bf16, float4 loads, clamped row (no
    //      predicated loads), pairs packed into b32/b64 LDS stores ----
#pragma unroll
    for (int i = 0; i < (GEMM_TR * KCH) / (256 * 4); ++i) {   // 4 iters
      int linear = i * 256 + tid;          // float4 index over 128x8
      int r  = linear >> 3;
      int c4 = linear & 7;
      int gr = rowBase + r;
      int cr = (gr < N) ? gr : (N - 1);    // always-legal address
      float4 v = reinterpret_cast<const float4*>(A + (size_t)cr * K + kc)[c4];
      if (gr >= N) { v.x = 0.0f; v.y = 0.0f; v.z = 0.0f; v.w = 0.0f; }
      unsigned lo = ((unsigned)f2bf(v.y) << 16) | f2bf(v.x);
      unsigned hi = ((unsigned)f2bf(v.w) << 16) | f2bf(v.z);
      unsigned* d = (unsigned*)&Alds[r][c4 * 4];
      d[0] = lo; d[1] = hi;
    }
    // Prefetch next A chunk (global_prefetch_b8 on gfx1250)
    if (kc + KCH < K) {
      int gr = rowBase + (tid & 127);
      if (gr < N)
        __builtin_prefetch(&A[(size_t)gr * K + kc + KCH + ((tid >> 7) * 16)], 0, 1);
    }
    __syncthreads();

    // ---- A fragment: lane holds row lr; VGPR j -> k=(j/8)*16+half*8+(j%8)
    BFrag a;
#pragma unroll
    for (int j = 0; j < 16; ++j) {
      int k = ((j >> 3) << 4) + (half << 3) + (j & 7);
      a.u[j] = Alds[wave * 16 + lr][k];
    }
#pragma unroll
    for (int ct = 0; ct < 4; ++ct)
      acc[ct] = __builtin_amdgcn_wmma_f32_16x16x32_bf16(
          false, a.v, false, bfr[ct].v, (short)0, acc[ct], false, false);
    __syncthreads();
  }

  // ---- write-out (C: VGPR r holds row half*8+r, lane&15 = col), fused bias
  const int waveRow = rowBase + wave * 16;
  const bool full = (waveRow + 16) <= N;
#pragma unroll
  for (int ct = 0; ct < 4; ++ct) {
    int nn = colBase + ct * 16 + lr;
    float bz = bias[nn];
    if (full) {
#pragma unroll
      for (int r = 0; r < 8; ++r)
        C[(size_t)(waveRow + half * 8 + r) * DH + nn] = acc[ct][r] + bz;
    } else {
#pragma unroll
      for (int r = 0; r < 8; ++r) {
        int row = waveRow + half * 8 + r;
        if (row < N) C[(size_t)row * DH + nn] = acc[ct][r] + bz;
      }
    }
  }
}

// ---------------------------------------------------------------------------
// out[i] = (1 + eps) * H[i]    (initializes aggregation buffer)
// ---------------------------------------------------------------------------
__global__ __launch_bounds__(256) void scale_copy(
    const float* __restrict__ H, const float* __restrict__ epsP,
    float* __restrict__ out, long n)
{
  long i = (long)blockIdx.x * 256 + threadIdx.x;
  if (i < n) out[i] = (1.0f + epsP[0]) * H[i];
}

// ---------------------------------------------------------------------------
// GIN sum aggregation: out[dst[e]] += H[src[e]]   (thread = edge x float4)
// ---------------------------------------------------------------------------
__global__ __launch_bounds__(256) void edge_scatter(
    const float* __restrict__ H, const int* __restrict__ src,
    const int* __restrict__ dst, float* __restrict__ out, int E)
{
  long idx = (long)blockIdx.x * 256 + threadIdx.x;
  if (idx >= (long)E * 64) return;
  int e  = (int)(idx >> 6);
  int c4 = (int)(idx & 63);
  const float4 v = reinterpret_cast<const float4*>(H + (long)src[e] * DH)[c4];
  float* o = out + (long)dst[e] * DH + (long)c4 * 4;
  unsafeAtomicAdd(o + 0, v.x);
  unsafeAtomicAdd(o + 1, v.y);
  unsafeAtomicAdd(o + 2, v.z);
  unsafeAtomicAdd(o + 3, v.w);
}

// ---------------------------------------------------------------------------
// Column sums / sums-of-squares over the node axis (for batch-norm stats).
// ---------------------------------------------------------------------------
__global__ __launch_bounds__(256) void col_stats(
    const float* __restrict__ X, int N,
    float* __restrict__ sums, float* __restrict__ sumsq)
{
  int col = threadIdx.x;
  int r0  = blockIdx.x * 128;
  float s = 0.0f, ss = 0.0f;
  int rend = (r0 + 128 < N) ? r0 + 128 : N;
  for (int r = r0; r < rend; ++r) {
    float v = X[(long)r * DH + col];
    s += v; ss += v * v;
  }
  unsafeAtomicAdd(&sums[col], s);
  unsafeAtomicAdd(&sumsq[col], ss);
}

// ---------------------------------------------------------------------------
// y = relu(g*(x-mean)*rsqrt(var+eps)+b); in-place, or accum[i] += y (residual)
// ---------------------------------------------------------------------------
__global__ __launch_bounds__(256) void bn_relu(
    float* __restrict__ X, const float* __restrict__ sums,
    const float* __restrict__ sumsq, const float* __restrict__ g,
    const float* __restrict__ b, int N, float* __restrict__ accum)
{
  long i = (long)blockIdx.x * 256 + threadIdx.x;
  if (i >= (long)N * DH) return;
  int col = (int)(i & (DH - 1));
  float inv  = 1.0f / (float)N;
  float mean = sums[col] * inv;
  float var  = sumsq[col] * inv - mean * mean;
  float sc   = g[col] * rsqrtf(var + 1e-5f);
  float v    = fmaxf(sc * (X[i] - mean) + b[col], 0.0f);
  if (accum) accum[i] += v;
  else       X[i] = v;
}

// ---------------------------------------------------------------------------
// Per-graph sum readout: out[gid[n]] += H[n]   (thread = node x float4)
// ---------------------------------------------------------------------------
__global__ __launch_bounds__(256) void readout_sum(
    const float* __restrict__ H, const int* __restrict__ gid,
    float* __restrict__ out, int N)
{
  long idx = (long)blockIdx.x * 256 + threadIdx.x;
  if (idx >= (long)N * 64) return;
  int n  = (int)(idx >> 6);
  int c4 = (int)(idx & 63);
  int gg = gid[n];
  const float4 v = reinterpret_cast<const float4*>(H + (long)n * DH)[c4];
  float* o = out + (long)gg * DH + (long)c4 * 4;
  unsafeAtomicAdd(o + 0, v.x);
  unsafeAtomicAdd(o + 1, v.y);
  unsafeAtomicAdd(o + 2, v.z);
  unsafeAtomicAdd(o + 3, v.w);
}

// ---------------------------------------------------------------------------
// y[g] = relu(|hg-hq| @ Wp1 + bp1) @ Wp2 + bp2   (64x256 -> 64, tiny)
// ---------------------------------------------------------------------------
__global__ __launch_bounds__(256) void predictor(
    const float* __restrict__ hg, const float* __restrict__ hq,
    const float* __restrict__ Wp1, const float* __restrict__ bp1,
    const float* __restrict__ Wp2, const float* __restrict__ bp2,
    float* __restrict__ y)
{
  __shared__ float red[256];
  int g = blockIdx.x;
  int j = threadIdx.x;
  float acc = bp1[j];
  for (int k = 0; k < DH; ++k) {
    float d = fabsf(hg[(long)g * DH + k] - hq[(long)g * DH + k]);
    acc += d * Wp1[(long)k * DH + j];
  }
  red[j] = fmaxf(acc, 0.0f) * Wp2[j];
  __syncthreads();
  for (int s = 128; s > 0; s >>= 1) {
    if (j < s) red[j] += red[j + s];
    __syncthreads();
  }
  if (j == 0) y[g] = red[0] + bp2[0];
}

// ===========================================================================
extern "C" void kernel_launch(void* const* d_in, const int* in_sizes, int n_in,
                              void* d_out, int out_size, void* d_ws, size_t ws_size,
                              hipStream_t stream)
{
  const float* X    = (const float*)d_in[0];
  const float* Xq   = (const float*)d_in[2];
  const float* Wemb = (const float*)d_in[4];
  const float* bemb = (const float*)d_in[5];
  const float* eps  = (const float*)d_in[6];
  const float* W1   = (const float*)d_in[7];
  const float* b1   = (const float*)d_in[8];
  const float* bn1g = (const float*)d_in[9];
  const float* bn1b = (const float*)d_in[10];
  const float* W2   = (const float*)d_in[11];
  const float* b2   = (const float*)d_in[12];
  const float* bnAg = (const float*)d_in[13];
  const float* bnAb = (const float*)d_in[14];
  const float* bnGg = (const float*)d_in[15];
  const float* bnGb = (const float*)d_in[16];
  const float* Wp1  = (const float*)d_in[17];
  const float* bp1  = (const float*)d_in[18];
  const float* Wp2  = (const float*)d_in[19];
  const float* bp2  = (const float*)d_in[20];
  const int* srcg = (const int*)d_in[21];
  const int* dstg = (const int*)d_in[22];
  const int* gidg = (const int*)d_in[23];
  const int* srcq = (const int*)d_in[24];
  const int* dstq = (const int*)d_in[25];
  const int* gidq = (const int*)d_in[26];

  const int DIN = 128;
  const int NG  = in_sizes[0] / DIN;
  const int NQ  = in_sizes[2] / DIN;
  const int EG  = in_sizes[21];
  const int EQ  = in_sizes[24];
  const int G   = 64;
  const int L   = 4;

  // ---- carve workspace (sized for the larger graph; reused for query) ----
  char* p = (char*)d_ws;
  auto carve = [&](size_t bytes) -> char* {
    char* r = p;
    p += (bytes + 255) & ~(size_t)255;
    return r;
  };
  float* bufH = (float*)carve((size_t)NG * DH * 4);   // residual H
  float* bufA = (float*)carve((size_t)NG * DH * 4);   // scratch A
  float* bufB = (float*)carve((size_t)NG * DH * 4);   // scratch B
  float* hg   = (float*)carve((size_t)G * DH * 4);
  float* hq   = (float*)carve((size_t)G * DH * 4);
  unsigned short* WembBf = (unsigned short*)carve((size_t)DIN * DH * 2);
  unsigned short* W1Bf   = (unsigned short*)carve((size_t)L * DH * DH * 2);
  unsigned short* W2Bf   = (unsigned short*)carve((size_t)L * DH * DH * 2);
  float* stats = (float*)carve(2 * DH * 4);           // sums | sumsq

  // ---- one-time per-launch weight pack: bf16, WMMA B-fragment order ----
  {
    long n0 = (long)DIN * DH;
    long n1 = (long)DH * DH;
    pack_weight_bf16<<<(unsigned)((n0 + 255) / 256), 256, 0, stream>>>(
        Wemb, WembBf, DIN);
    for (int l = 0; l < L; ++l) {
      pack_weight_bf16<<<(unsigned)((n1 + 255) / 256), 256, 0, stream>>>(
          W1 + (size_t)l * DH * DH, W1Bf + (size_t)l * DH * DH, DH);
      pack_weight_bf16<<<(unsigned)((n1 + 255) / 256), 256, 0, stream>>>(
          W2 + (size_t)l * DH * DH, W2Bf + (size_t)l * DH * DH, DH);
    }
  }

  auto run_graph = [&](const float* Xin, int N, const int* src, const int* dst,
                       int E, const int* gid, float* hout) {
    const unsigned gx  = (unsigned)((N + GEMM_TR - 1) / GEMM_TR);
    const long     n   = (long)N * DH;
    const unsigned gel = (unsigned)((n + 255) / 256);
    const unsigned gsc = (unsigned)(((long)E * 64 + 255) / 256);
    const unsigned gst = (unsigned)((N + 127) / 128);

    // H = X @ W_emb + b_emb
    gemm_wmma_bf16<<<dim3(gx, DH / GEMM_TC), 256, 0, stream>>>(
        Xin, WembBf, bemb, bufH, N, DIN);

    for (int l = 0; l < L; ++l) {
      // h0 = (1+eps)*H + segment_sum(H[src] -> dst)
      scale_copy<<<gel, 256, 0, stream>>>(bufH, eps + l, bufA, n);
      edge_scatter<<<gsc, 256, 0, stream>>>(bufH, src, dst, bufA, E);

      // t1 = relu(bn1(h0 @ W1 + b1))
      gemm_wmma_bf16<<<dim3(gx, DH / GEMM_TC), 256, 0, stream>>>(
          bufA, W1Bf + (size_t)l * DH * DH, b1 + (size_t)l * DH, bufB, N, DH);
      hipMemsetAsync(stats, 0, 2 * DH * 4, stream);
      col_stats<<<gst, 256, 0, stream>>>(bufB, N, stats, stats + DH);
      bn_relu<<<gel, 256, 0, stream>>>(bufB, stats, stats + DH,
                                       bn1g + (size_t)l * DH, bn1b + (size_t)l * DH,
                                       N, nullptr);

      // t2 = t1 @ W2 + b2 ; relu(bnA) ; H += relu(bnG)
      gemm_wmma_bf16<<<dim3(gx, DH / GEMM_TC), 256, 0, stream>>>(
          bufB, W2Bf + (size_t)l * DH * DH, b2 + (size_t)l * DH, bufA, N, DH);
      hipMemsetAsync(stats, 0, 2 * DH * 4, stream);
      col_stats<<<gst, 256, 0, stream>>>(bufA, N, stats, stats + DH);
      bn_relu<<<gel, 256, 0, stream>>>(bufA, stats, stats + DH,
                                       bnAg + (size_t)l * DH, bnAb + (size_t)l * DH,
                                       N, nullptr);
      hipMemsetAsync(stats, 0, 2 * DH * 4, stream);
      col_stats<<<gst, 256, 0, stream>>>(bufA, N, stats, stats + DH);
      bn_relu<<<gel, 256, 0, stream>>>(bufA, stats, stats + DH,
                                       bnGg + (size_t)l * DH, bnGb + (size_t)l * DH,
                                       N, bufH);   // residual: H += relu(bnG(t2))
    }

    // per-graph sum readout
    hipMemsetAsync(hout, 0, (size_t)G * DH * 4, stream);
    unsigned gro = (unsigned)(((long)N * 64 + 255) / 256);
    readout_sum<<<gro, 256, 0, stream>>>(bufH, gid, hout, N);
  };

  run_graph(X,  NG, srcg, dstg, EG, gidg, hg);
  run_graph(Xq, NQ, srcq, dstq, EQ, gidq, hq);

  predictor<<<G, 256, 0, stream>>>(hg, hq, Wp1, bp1, Wp2, bp2, (float*)d_out);
}
